// Rnn_28020366639738
// MI455X (gfx1250) — matrix operand order repaired and checked by reading, not compiled, not run on hardware
//
#include <hip/hip_runtime.h>

// Problem constants (match reference)
constexpr int kH = 256;   // hidden per direction
constexpr int kI = 50;    // input size
constexpr int kB = 128;   // batch
constexpr int kT = 512;   // sequence length

typedef _Float16 v16h __attribute__((ext_vector_type(16)));
typedef _Float16 v8h  __attribute__((ext_vector_type(8)));
typedef float    v8f  __attribute__((ext_vector_type(8)));
typedef float    v4f  __attribute__((ext_vector_type(4)));

static __device__ __forceinline__ v16h combine8(v8h lo, v8h hi) {
    return __builtin_shufflevector(lo, hi, 0, 1, 2, 3, 4, 5, 6, 7,
                                           8, 9, 10, 11, 12, 13, 14, 15);
}

// ---------------------------------------------------------------------------
// Kernel 1: input projection for both directions, written straight into d_out
//   out[b][t][h]      = x[b,t,:] . w_ih_f[h,:] + b_ih_f[h] + b_hh_f[h]
//   out[b][t][256+h]  = x[b,t,:] . w_ih_b[h,:] + b_ih_b[h] + b_hh_b[h]
// (backward projection stored at ORIGINAL time index; the scan walks t in
//  reverse for dir=1, which reproduces reference xp_b ordering.)
// ---------------------------------------------------------------------------
__global__ __launch_bounds__(256) void rnn_proj(
    const float* __restrict__ x,
    const float* __restrict__ w_f, const float* __restrict__ bih_f,
    const float* __restrict__ bhh_f,
    const float* __restrict__ w_b, const float* __restrict__ bih_b,
    const float* __restrict__ bhh_b,
    float* __restrict__ out)
{
    __shared__ float xs[8][kI + 2];
    const int b  = blockIdx.y;
    const int t0 = blockIdx.x * 8;

    for (int idx = threadIdx.x; idx < 8 * kI; idx += 256) {
        const int j = idx / kI, i = idx % kI;
        xs[j][i] = x[((size_t)b * kT + (t0 + j)) * kI + i];
    }
    __syncthreads();

    const int h = threadIdx.x;   // 0..255
    float wf[kI], wb[kI];
#pragma unroll
    for (int i = 0; i < kI; ++i) wf[i] = w_f[h * kI + i];
#pragma unroll
    for (int i = 0; i < kI; ++i) wb[i] = w_b[h * kI + i];
    const float cf = bih_f[h] + bhh_f[h];
    const float cb = bih_b[h] + bhh_b[h];

    for (int j = 0; j < 8; ++j) {
        float af = cf, ab = cb;
#pragma unroll
        for (int i = 0; i < kI; ++i) {
            const float xv = xs[j][i];
            af = fmaf(xv, wf[i], af);
            ab = fmaf(xv, wb[i], ab);
        }
        const size_t o = ((size_t)b * kT + (t0 + j)) * (2 * kH) + h;
        out[o]      = af;
        out[o + kH] = ab;
    }
}

// ---------------------------------------------------------------------------
// Kernel 2: recurrent scan.  grid = (B/16, 2 dirs), 256 threads = 8 waves.
// Each workgroup owns 16 batch rows for one direction; h state (16x256 f16)
// ping-pongs between two LDS buffers (single barrier per step); W_hh lives
// entirely in registers as pre-formatted WMMA B-fragments (f16).
// Per step: h_t = relu(xp_t + h_{t-1} @ W_hh^T), xp read from / h written to
// d_out in place.
// ---------------------------------------------------------------------------
__global__ __launch_bounds__(256) void rnn_scan(
    const float* __restrict__ whh_f,
    const float* __restrict__ whh_b,
    float* __restrict__ out)
{
    __shared__ _Float16 hping[16 * kH];         // 8 KB each: h tile (16 x 256)
    __shared__ _Float16 hpong[16 * kH];

    const int dir  = blockIdx.y;
    const int b0   = blockIdx.x * 16;
    const int lane = threadIdx.x & 31;
    const int wv   = threadIdx.x >> 5;          // wave id 0..7
    const int m16  = lane & 15;
    const int hi16 = lane >> 4;                 // 0 or 1 (half-wave)
    const float* __restrict__ whh = dir ? whh_b : whh_f;

    // ---- Hoist B = W_hh^T fragments into registers (f16), 2 ntiles x 8 ksteps.
    // B (32x16, 16-bit): lane l holds column n = l%16, K = ks*32 + (l<16?0:16)
    // .. +15 consecutive. Column n of W^T == row n of W -> contiguous loads.
    v16h Bf[2][8];
#pragma unroll
    for (int nt = 0; nt < 2; ++nt) {
        const int n = wv * 32 + nt * 16 + m16;
#pragma unroll
        for (int ks = 0; ks < 8; ++ks) {
            const int kb = ks * 32 + hi16 * 16;
            const v4f* wr = (const v4f*)(whh + (size_t)n * kH + kb);
            const v4f q0 = wr[0], q1 = wr[1], q2 = wr[2], q3 = wr[3];
            v16h f;
#pragma unroll
            for (int j = 0; j < 4; ++j) {
                f[j]      = (_Float16)q0[j];
                f[4 + j]  = (_Float16)q1[j];
                f[8 + j]  = (_Float16)q2[j];
                f[12 + j] = (_Float16)q3[j];
            }
            Bf[nt][ks] = f;
        }
    }

    // ---- h0 = 0 (read buffer of step 0)
    for (int idx = threadIdx.x; idx < 16 * kH; idx += 256)
        hping[idx] = (_Float16)0.0f;

    const int kofA = hi16 * 16;                 // A-fragment K base offset
    const int colb = wv * 32;                   // this wave's local column base

    const _Float16* rb = hping;                 // read h_{t-1} from here
    _Float16*       wb = hpong;                 // write h_t here

    for (int t = 0; t < kT; ++t) {
        const int tt = dir ? (kT - 1 - t) : t;
        // Single barrier per step: makes h_{t-1} writes (into rb) visible AND
        // guarantees everyone finished reading wb during step t-1 (wb was the
        // read buffer then), so we may overwrite it below.
        __syncthreads();

        // Prefetch xp for this step (independent of h -> hides HBM latency
        // behind the WMMA chain below).
        float xp[2][8];
#pragma unroll
        for (int nt = 0; nt < 2; ++nt) {
            const size_t cb = (size_t)dir * kH + colb + nt * 16 + m16;
#pragma unroll
            for (int r = 0; r < 8; ++r) {
                const int m = r + hi16 * 8;
                xp[nt][r] = out[((size_t)(b0 + m) * kT + tt) * (2 * kH) + cb];
            }
        }

        v8f acc0 = {0.f, 0.f, 0.f, 0.f, 0.f, 0.f, 0.f, 0.f};
        v8f acc1 = acc0;
#pragma unroll
        for (int ks = 0; ks < 8; ++ks) {
            // A (16x32, 16-bit): lane l holds row m=l%16,
            // K = ks*32 + (l<16?0:16) .. +15 consecutive.
            const int ka = ks * 32 + kofA;
            const v8h lo = *(const v8h*)&rb[m16 * kH + ka];
            const v8h hi = *(const v8h*)&rb[m16 * kH + ka + 8];
            const v16h a = combine8(lo, hi);
            acc0 = __builtin_amdgcn_wmma_f32_16x16x32_f16(
                       false, a, false, Bf[0][ks], (short)0, acc0, false, false);
            acc1 = __builtin_amdgcn_wmma_f32_16x16x32_f16(
                       false, a, false, Bf[1][ks], (short)0, acc1, false, false);
        }

        // Epilogue: h_t = relu(acc + xp); write f32 to global (in place over
        // xp) and f16 to the ping-pong write buffer for the next step.
#pragma unroll
        for (int nt = 0; nt < 2; ++nt) {
            const v8f acc = nt ? acc1 : acc0;
            const int cl = colb + nt * 16 + m16;
#pragma unroll
            for (int r = 0; r < 8; ++r) {
                const int m = r + hi16 * 8;
                float v = acc[r] + xp[nt][r];
                v = fmaxf(v, 0.0f);
                out[((size_t)(b0 + m) * kT + tt) * (2 * kH) + dir * kH + cl] = v;
                wb[m * kH + cl] = (_Float16)v;
            }
        }

        // swap ping-pong buffers
        const _Float16* tr = rb;
        rb = wb;
        wb = (_Float16*)tr;
    }
}

// ---------------------------------------------------------------------------
extern "C" void kernel_launch(void* const* d_in, const int* in_sizes, int n_in,
                              void* d_out, int out_size, void* d_ws, size_t ws_size,
                              hipStream_t stream) {
    (void)in_sizes; (void)n_in; (void)out_size; (void)d_ws; (void)ws_size;
    const float* x    = (const float*)d_in[0];
    const float* wihf = (const float*)d_in[1];
    const float* whhf = (const float*)d_in[2];
    const float* bihf = (const float*)d_in[3];
    const float* bhhf = (const float*)d_in[4];
    const float* wihb = (const float*)d_in[5];
    const float* whhb = (const float*)d_in[6];
    const float* bihb = (const float*)d_in[7];
    const float* bhhb = (const float*)d_in[8];
    float* out = (float*)d_out;

    // 1) xp (input projection + biases) -> d_out, both directions.
    rnn_proj<<<dim3(kT / 8, kB), 256, 0, stream>>>(
        x, wihf, bihf, bhhf, wihb, bihb, bhhb, out);

    // 2) sequential scan, in place over d_out, one barrier per time step.
    rnn_scan<<<dim3(kB / 16, 2), 256, 0, stream>>>(whhf, whhb, out);
}
